// SelfAttention_1314259992903
// MI455X (gfx1250) — matrix operand order repaired and checked by reading, not compiled, not run on hardware
//
#include <hip/hip_runtime.h>
#include <hip/hip_bf16.h>

// ---------------------------------------------------------------------------
// Types for CDNA5 WMMA (wave32, 16x16x32 bf16 -> f32)
// ---------------------------------------------------------------------------
typedef __attribute__((ext_vector_type(16))) __bf16        bf16x16;
typedef __attribute__((ext_vector_type(8)))  __bf16        bf16x8;
typedef __attribute__((ext_vector_type(8)))  float         f32x8;
typedef __attribute__((ext_vector_type(4)))  unsigned int  u32x4;

#define WMMA_BF16(a, b, c)                                                     \
  __builtin_amdgcn_wmma_f32_16x16x32_bf16(false, (a), false, (b), (short)0,    \
                                          (c), false, false)

// gfx1250 async global->LDS path (ASYNCcnt), if the toolchain exposes it.
#if defined(__has_builtin)
#if __has_builtin(__builtin_amdgcn_global_load_async_to_lds_b128) && \
    __has_builtin(__builtin_amdgcn_s_wait_asynccnt)
#define USE_ASYNC_LDS 1
#endif
#endif

#ifdef USE_ASYNC_LDS
// Builtin signature (from clang diagnostic): first param is
// 'int __vector(4) __device__ *' i.e. v4i in addrspace(1); LDS side is v4i in
// addrspace(3); then two integer immediates (offset, cpol).
typedef int v4i_vs __attribute__((vector_size(16)));
typedef __attribute__((address_space(1))) v4i_vs as1_v4i;
typedef __attribute__((address_space(3))) v4i_vs as3_v4i;

static __device__ __forceinline__ void async_b128(const __bf16* g, __bf16* l) {
  __builtin_amdgcn_global_load_async_to_lds_b128(
      (as1_v4i*)g, (as3_v4i*)l, /*imm offset*/ 0, /*cpol*/ 0);
}
#endif

// Build a 16-element A/B fragment from two contiguous 16-byte halves.
static __device__ __forceinline__ bf16x16 frag2(const __bf16* p0,
                                                const __bf16* p1) {
  bf16x8 a = *(const bf16x8*)p0;
  bf16x8 b = *(const bf16x8*)p1;
  bf16x16 r;
#pragma unroll
  for (int i = 0; i < 8; ++i) { r[i] = a[i]; r[i + 8] = b[i]; }
  return r;
}

static __device__ __forceinline__ f32x8 zero8() {
  f32x8 z = {0.f, 0.f, 0.f, 0.f, 0.f, 0.f, 0.f, 0.f};
  return z;
}

// ---------------------------------------------------------------------------
// Problem constants
// ---------------------------------------------------------------------------
#define BATCH 2
#define SEQ   2048
#define DIM   1024
#define HEADS 16
#define DH    64
#define INNER 1024
#define N3    3072
#define ROWS  (BATCH * SEQ)      // 4096
#define CEXP  0.180336880f       // SCALE * log2(e) = 64^-0.5 * 1.4426950
#define EPS   1e-10f

#define LDSP  40                 // LDS row stride in bf16 (80B = 20 banks, conflict-free)
#define BUFSZ (128 * LDSP)       // one stage of one operand tile

// ---------------------------------------------------------------------------
// fp32 -> bf16 elementwise convert (x)
// ---------------------------------------------------------------------------
typedef __attribute__((ext_vector_type(8))) float f32v8;

__global__ __launch_bounds__(256) void k_cvt(const float* __restrict__ in,
                                             __bf16* __restrict__ out, int n) {
  int i = (blockIdx.x * 256 + threadIdx.x) * 8;
  if (i + 8 <= n) {
    f32v8 x = *(const f32v8*)(in + i);
    bf16x8 v;
#pragma unroll
    for (int j = 0; j < 8; ++j) v[j] = (__bf16)x[j];
    *(bf16x8*)(out + i) = v;
  }
}

// fp32 [R][C] -> bf16 transposed [C][R]   (R = gridDim.y)
__global__ __launch_bounds__(256) void k_cvt_t(const float* __restrict__ in,
                                               __bf16* __restrict__ out, int C) {
  int r = blockIdx.y;
  int c = blockIdx.x * 256 + threadIdx.x;
  int R = gridDim.y;
  out[(size_t)c * R + r] = (__bf16)in[(size_t)r * C + c];
}

// ---------------------------------------------------------------------------
// Shared GEMM mainloop: C[128x128] += A[128xK] * Bt[128xK]^T, K=1024, bf16->f32
// 8 waves (2x4), each wave owns a 64x32 sub-tile = 4x2 WMMA accumulators.
// sa/sb are double-buffered (2*BUFSZ each) when the async path is available.
// ---------------------------------------------------------------------------
static __device__ __forceinline__ void gemm_main(
    const __bf16* __restrict__ A, const __bf16* __restrict__ Bt, int m0, int n0,
    __bf16* sa, __bf16* sb, f32x8 acc[4][2]) {
  const int tid  = threadIdx.x;
  const int wave = tid >> 5, lane = tid & 31;
  const int wm = wave >> 2, wn = wave & 3;
  const int r = lane & 15, hi = lane >> 4;
  const int srow = tid >> 1, sseg = (tid & 1) * 16;

  const __bf16* agp = A + (size_t)(m0 + srow) * 1024 + sseg;
  const __bf16* bgp = Bt + (size_t)(n0 + srow) * 1024 + sseg;
  __bf16* asp = sa + srow * LDSP + sseg;
  __bf16* bsp = sb + srow * LDSP + sseg;

#ifdef USE_ASYNC_LDS
  // ---- double-buffered: async tile t+1 overlaps WMMAs on tile t ----
#define STAGE_ASYNC(buf, koff)                                                 \
  do {                                                                         \
    async_b128(agp + (koff),     asp + (buf)*BUFSZ);                           \
    async_b128(agp + (koff) + 8, asp + (buf)*BUFSZ + 8);                       \
    async_b128(bgp + (koff),     bsp + (buf)*BUFSZ);                           \
    async_b128(bgp + (koff) + 8, bsp + (buf)*BUFSZ + 8);                       \
  } while (0)

  STAGE_ASYNC(0, 0);
  for (int t = 0; t < 32; ++t) {
    const int cur = t & 1;
    if (t + 1 < 32) {
      STAGE_ASYNC(1 - cur, (t + 1) * 32);
      __builtin_amdgcn_s_wait_asynccnt(4);  // in-order: oldest 4 (tile t) done
    } else {
      __builtin_amdgcn_s_wait_asynccnt(0);
    }
    __syncthreads();                         // tile t visible to all waves

    bf16x16 af[4], bfr[2];
#pragma unroll
    for (int i = 0; i < 4; ++i) {
      const __bf16* p = sa + cur * BUFSZ + (wm * 64 + i * 16 + r) * LDSP;
      af[i] = frag2(p + 8 * hi, p + 16 + 8 * hi);
    }
#pragma unroll
    for (int j = 0; j < 2; ++j) {
      const __bf16* p = sb + cur * BUFSZ + (wn * 32 + j * 16 + r) * LDSP;
      bfr[j] = frag2(p + 8 * hi, p + 16 + 8 * hi);
    }
#pragma unroll
    for (int i = 0; i < 4; ++i)
#pragma unroll
      for (int j = 0; j < 2; ++j)
        acc[i][j] = WMMA_BF16(af[i], bfr[j], acc[i][j]);
    __syncthreads();                         // reads done before buf reuse
  }
#undef STAGE_ASYNC
#else
  // ---- fallback: synchronous VGPR staging, single buffer ----
  for (int k = 0; k < 1024; k += 32) {
    u32x4 a0 = *(const u32x4*)(agp + k);
    u32x4 a1 = *(const u32x4*)(agp + k + 8);
    u32x4 b0 = *(const u32x4*)(bgp + k);
    u32x4 b1 = *(const u32x4*)(bgp + k + 8);
    __syncthreads();
    *(u32x4*)asp = a0;       *(u32x4*)(asp + 8) = a1;
    *(u32x4*)bsp = b0;       *(u32x4*)(bsp + 8) = b1;
    __syncthreads();

    bf16x16 af[4], bfr[2];
#pragma unroll
    for (int i = 0; i < 4; ++i) {
      const __bf16* p = sa + (wm * 64 + i * 16 + r) * LDSP;
      af[i] = frag2(p + 8 * hi, p + 16 + 8 * hi);
    }
#pragma unroll
    for (int j = 0; j < 2; ++j) {
      const __bf16* p = sb + (wn * 32 + j * 16 + r) * LDSP;
      bfr[j] = frag2(p + 8 * hi, p + 16 + 8 * hi);
    }
#pragma unroll
    for (int i = 0; i < 4; ++i)
#pragma unroll
      for (int j = 0; j < 2; ++j)
        acc[i][j] = WMMA_BF16(af[i], bfr[j], acc[i][j]);
  }
#endif
}

// ---------------------------------------------------------------------------
// Kernel 1: QKV = Xbf @ Wqkv_t^T ; scatter into Q[bh][l][dh], K[bh][l][dh],
//           Vt[bh][dh][l].  s = n0>>10 is BLOCK-UNIFORM (N-tiles never cross
//           the Q/K/V boundary), so the 3-way select is hoisted out of the
//           per-element loops.
// ---------------------------------------------------------------------------
__global__ __launch_bounds__(256) void k_gemm_qkv(
    const __bf16* __restrict__ A, const __bf16* __restrict__ Bt,
    __bf16* __restrict__ Qb, __bf16* __restrict__ Kb, __bf16* __restrict__ Vt) {
  __shared__ __bf16 sa[2 * BUFSZ];
  __shared__ __bf16 sb[2 * BUFSZ];
  const int m0 = blockIdx.y * 128, n0 = blockIdx.x * 128;
  f32x8 acc[4][2];
#pragma unroll
  for (int i = 0; i < 4; ++i)
#pragma unroll
    for (int j = 0; j < 2; ++j) acc[i][j] = zero8();

  gemm_main(A, Bt, m0, n0, sa, sb, acc);

  const int tid = threadIdx.x;
  const int wave = tid >> 5, lane = tid & 31;
  const int wm = wave >> 2, wn = wave & 3;
  const int r = lane & 15, hi = lane >> 4;
  const int s = n0 >> 10;                   // 0=Q, 1=K, 2=V (uniform)

  if (s == 2) {                             // V: store transposed [bh][dh][l]
#pragma unroll
    for (int i = 0; i < 4; ++i) {
#pragma unroll
      for (int j = 0; j < 2; ++j) {
        const int gcol = n0 + wn * 32 + j * 16 + r;
        const int hh = (gcol >> 6) & 15, dd = gcol & 63;
#pragma unroll
        for (int v = 0; v < 8; ++v) {
          const int gr = m0 + wm * 64 + i * 16 + v + 8 * hi;
          const int b = gr >> 11, q = gr & 2047;
          Vt[((size_t)(b * HEADS + hh) * DH + dd) * SEQ + q] =
              (__bf16)acc[i][j][v];
        }
      }
    }
  } else {                                  // Q or K: row-major [bh][l][dh]
    __bf16* dst = s ? Kb : Qb;
#pragma unroll
    for (int i = 0; i < 4; ++i) {
#pragma unroll
      for (int j = 0; j < 2; ++j) {
        const int gcol = n0 + wn * 32 + j * 16 + r;
        const int hh = (gcol >> 6) & 15, dd = gcol & 63;
#pragma unroll
        for (int v = 0; v < 8; ++v) {
          const int gr = m0 + wm * 64 + i * 16 + v + 8 * hi;
          const int b = gr >> 11, q = gr & 2047;
          dst[((size_t)(b * HEADS + hh) * SEQ + q) * DH + dd] =
              (__bf16)acc[i][j][v];
        }
      }
    }
  }
}

// ---------------------------------------------------------------------------
// Kernel 2: flash attention with exact post-softmax multiplicative mask.
//   attn_i = e^{s_i - M} mask_i / (sum_j e^{s_j - M} mask_j + EPS * Z)
// Track running max, masked sum, and unmasked sum (Z) per row.
// 1 wave = 16 query rows; block = 8 waves = 128 queries; grid (16, b*h=32).
// ---------------------------------------------------------------------------
__global__ __launch_bounds__(256) void k_attn(
    const __bf16* __restrict__ Qb, const __bf16* __restrict__ Kb,
    const __bf16* __restrict__ Vt, const float* __restrict__ mask,
    __bf16* __restrict__ Zb) {
  __shared__ __bf16 plds[8][16][LDSP];   // per-wave P tile [16 q][32 keys]

  const int tid = threadIdx.x;
  const int wave = tid >> 5, lane = tid & 31;
  const int r = lane & 15, hi = lane >> 4;
  const int bh = blockIdx.y;
  const int b = bh >> 4, h = bh & 15;
  const int q0 = blockIdx.x * 128 + wave * 16;

  const __bf16* Q  = Qb + (size_t)bh * SEQ * DH;
  const __bf16* Kp = Kb + (size_t)bh * SEQ * DH;
  const __bf16* Vp = Vt + (size_t)bh * DH * SEQ;
  const float*  Mb = mask + (size_t)b * SEQ * SEQ;

  // Resident Q A-fragments (dh = 64 -> two 16x32 fragments)
  bf16x16 qf0, qf1;
  {
    const __bf16* base = Q + (size_t)(q0 + r) * DH;
    qf0 = frag2(base + 8 * hi, base + 16 + 8 * hi);
    qf1 = frag2(base + 32 + 8 * hi, base + 48 + 8 * hi);
  }

  f32x8 acc[4];
#pragma unroll
  for (int nt = 0; nt < 4; ++nt) acc[nt] = zero8();
  float m_run[8], zsum[8], msum[8];
#pragma unroll
  for (int v = 0; v < 8; ++v) { m_run[v] = -1e30f; zsum[v] = 0.f; msum[v] = 0.f; }

  for (int n0 = 0; n0 < SEQ; n0 += 32) {
    // prefetch next chunk's K rows / V rows into near caches (L2-resident)
    if (n0 + 32 < SEQ) {
      __builtin_prefetch(Kp + (size_t)(n0 + 32 + r) * DH, 0, 3);
      __builtin_prefetch(Kp + (size_t)(n0 + 48 + r) * DH, 0, 3);
      __builtin_prefetch(Vp + (size_t)r * SEQ + n0 + 32, 0, 3);
      __builtin_prefetch(Mb + (size_t)(q0 + r) * SEQ + n0 + 32, 0, 3);
    }

    // ---- scores: two 16x16 tiles (keys n0..+15, n0+16..+31), K-dim = 64 ----
    f32x8 s0 = zero8(), s1 = zero8();
    {
      const __bf16* k0p = Kp + (size_t)(n0 + r) * DH;
      const __bf16* k1p = Kp + (size_t)(n0 + 16 + r) * DH;
      bf16x16 kb00 = frag2(k0p + 8 * hi,      k0p + 16 + 8 * hi);
      bf16x16 kb01 = frag2(k0p + 32 + 8 * hi, k0p + 48 + 8 * hi);
      bf16x16 kb10 = frag2(k1p + 8 * hi,      k1p + 16 + 8 * hi);
      bf16x16 kb11 = frag2(k1p + 32 + 8 * hi, k1p + 48 + 8 * hi);
      s0 = WMMA_BF16(qf0, kb00, s0);
      s0 = WMMA_BF16(qf1, kb01, s0);
      s1 = WMMA_BF16(qf0, kb10, s1);
      s1 = WMMA_BF16(qf1, kb11, s1);
    }

    // ---- multiplicative mask (lane's column = key n0+r / n0+16+r) ----
    float mk0[8], mk1[8];
#pragma unroll
    for (int v = 0; v < 8; ++v) {
      const float* mr = Mb + (size_t)(q0 + v + 8 * hi) * SEQ + n0;
      mk0[v] = mr[r];
      mk1[v] = mr[16 + r];
    }

    // ---- online softmax stats per row (butterfly over 16-lane group) ----
    float p0[8], p1[8], alpha[8];
#pragma unroll
    for (int v = 0; v < 8; ++v) {
      float cm = fmaxf(s0[v], s1[v]);
      cm = fmaxf(cm, __shfl_xor(cm, 1, 32));
      cm = fmaxf(cm, __shfl_xor(cm, 2, 32));
      cm = fmaxf(cm, __shfl_xor(cm, 4, 32));
      cm = fmaxf(cm, __shfl_xor(cm, 8, 32));
      float mn = fmaxf(m_run[v], cm);
      float a  = exp2f((m_run[v] - mn) * CEXP);
      m_run[v] = mn;
      float e0 = exp2f((s0[v] - mn) * CEXP);
      float e1 = exp2f((s1[v] - mn) * CEXP);
      float pm0 = e0 * mk0[v], pm1 = e1 * mk1[v];
      float zl = e0 + e1;
      float ml = pm0 + pm1;
      zl += __shfl_xor(zl, 1, 32); zl += __shfl_xor(zl, 2, 32);
      zl += __shfl_xor(zl, 4, 32); zl += __shfl_xor(zl, 8, 32);
      ml += __shfl_xor(ml, 1, 32); ml += __shfl_xor(ml, 2, 32);
      ml += __shfl_xor(ml, 4, 32); ml += __shfl_xor(ml, 8, 32);
      zsum[v] = zsum[v] * a + zl;
      msum[v] = msum[v] * a + ml;
      alpha[v] = a;
      p0[v] = pm0; p1[v] = pm1;
    }

    // ---- rescale accumulators ----
#pragma unroll
    for (int nt = 0; nt < 4; ++nt)
#pragma unroll
      for (int v = 0; v < 8; ++v) acc[nt][v] *= alpha[v];

    // ---- C-layout -> A-layout transpose of P via private LDS tile ----
#pragma unroll
    for (int v = 0; v < 8; ++v) {
      plds[wave][v + 8 * hi][r]      = (__bf16)p0[v];
      plds[wave][v + 8 * hi][16 + r] = (__bf16)p1[v];
    }
    __syncthreads();
    bf16x16 pf;
    {
      const __bf16* pr = &plds[wave][r][0];
      pf = frag2(pr + 8 * hi, pr + 16 + 8 * hi);
    }

    // ---- acc += P @ V  (Vt gives contiguous B-fragments) ----
#pragma unroll
    for (int nt = 0; nt < 4; ++nt) {
      const __bf16* vp = Vp + (size_t)(nt * 16 + r) * SEQ + n0 + 8 * hi;
      bf16x16 vf = frag2(vp, vp + 16);
      acc[nt] = WMMA_BF16(pf, vf, acc[nt]);
    }
    __syncthreads();
  }

  // ---- renormalize exactly as reference and store z as bf16 [b*l][inner] ---
#pragma unroll
  for (int v = 0; v < 8; ++v) {
    const float inv = 1.0f / (msum[v] + EPS * zsum[v]);
    const int q = q0 + v + 8 * hi;
#pragma unroll
    for (int nt = 0; nt < 4; ++nt) {
      const int col = h * DH + nt * 16 + r;
      Zb[(size_t)(b * SEQ + q) * INNER + col] = (__bf16)(acc[nt][v] * inv);
    }
  }
}

// ---------------------------------------------------------------------------
// Kernel 3: out = Z @ Wout_t^T + b_out  (fp32 output)
// ---------------------------------------------------------------------------
__global__ __launch_bounds__(256) void k_gemm_out(
    const __bf16* __restrict__ A, const __bf16* __restrict__ Bt,
    const float* __restrict__ bias, float* __restrict__ out) {
  __shared__ __bf16 sa[2 * BUFSZ];
  __shared__ __bf16 sb[2 * BUFSZ];
  const int m0 = blockIdx.y * 128, n0 = blockIdx.x * 128;
  f32x8 acc[4][2];
#pragma unroll
  for (int i = 0; i < 4; ++i)
#pragma unroll
    for (int j = 0; j < 2; ++j) acc[i][j] = zero8();

  gemm_main(A, Bt, m0, n0, sa, sb, acc);

  const int tid = threadIdx.x;
  const int wave = tid >> 5, lane = tid & 31;
  const int wm = wave >> 2, wn = wave & 3;
  const int r = lane & 15, hi = lane >> 4;
#pragma unroll
  for (int i = 0; i < 4; ++i) {
#pragma unroll
    for (int j = 0; j < 2; ++j) {
      const int gcol = n0 + wn * 32 + j * 16 + r;
      const float bj = bias[gcol];
#pragma unroll
      for (int v = 0; v < 8; ++v) {
        const int gr = m0 + wm * 64 + i * 16 + v + 8 * hi;
        out[(size_t)gr * DIM + gcol] = acc[i][j][v] + bj;
      }
    }
  }
}

// ---------------------------------------------------------------------------
// Host-side launcher
// ---------------------------------------------------------------------------
extern "C" void kernel_launch(void* const* d_in, const int* in_sizes, int n_in,
                              void* d_out, int out_size, void* d_ws,
                              size_t ws_size, hipStream_t stream) {
  (void)in_sizes; (void)n_in; (void)out_size; (void)ws_size;
  const float* x    = (const float*)d_in[0];
  const float* mask = (const float*)d_in[1];
  const float* Wqkv = (const float*)d_in[2];
  const float* Wout = (const float*)d_in[3];
  const float* bout = (const float*)d_in[4];
  float* out = (float*)d_out;

  // Workspace carve-out (all bf16), total ~48 MB
  char* ws = (char*)d_ws;
  size_t o = 0;
  __bf16* Xbf = (__bf16*)(ws + o); o += (size_t)ROWS * DIM * 2;      // 8 MB
  __bf16* Wqt = (__bf16*)(ws + o); o += (size_t)N3 * DIM * 2;        // 6 MB
  __bf16* Wot = (__bf16*)(ws + o); o += (size_t)DIM * INNER * 2;     // 2 MB
  __bf16* Qb  = (__bf16*)(ws + o); o += (size_t)BATCH * HEADS * SEQ * DH * 2;
  __bf16* Kb  = (__bf16*)(ws + o); o += (size_t)BATCH * HEADS * SEQ * DH * 2;
  __bf16* Vt  = (__bf16*)(ws + o); o += (size_t)BATCH * HEADS * DH * SEQ * 2;
  __bf16* Zb  = (__bf16*)(ws + o); o += (size_t)ROWS * INNER * 2;

  // 0) one-time precision conversion / weight transposes (stays in L2)
  k_cvt<<<(ROWS * DIM) / (8 * 256), 256, 0, stream>>>(x, Xbf, ROWS * DIM);
  k_cvt_t<<<dim3(N3 / 256, DIM), 256, 0, stream>>>(Wqkv, Wqt, N3);
  k_cvt_t<<<dim3(INNER / 256, DIM), 256, 0, stream>>>(Wout, Wot, INNER);

  // 1) QKV projection (scatters Q, K, V-transposed per head)
  k_gemm_qkv<<<dim3(N3 / 128, ROWS / 128), 256, 0, stream>>>(Xbf, Wqt, Qb, Kb, Vt);

  // 2) flash attention with post-softmax multiplicative mask (exact renorm)
  k_attn<<<dim3(SEQ / 128, BATCH * HEADS), 256, 0, stream>>>(Qb, Kb, Vt, mask, Zb);

  // 3) output projection + bias
  k_gemm_out<<<dim3(DIM / 128, ROWS / 128), 256, 0, stream>>>(Zb, Wot, bout, out);
}
